// GAT_nl_72464688218547
// MI455X (gfx1250) — matrix operand order repaired and checked by reading, not compiled, not run on hardware
//
#include <hip/hip_runtime.h>

#define NN 8192
#define FIN 256
#define FOUT 64
#define ALPHA 0.2f
#define LOG2E 1.4426950408889634f
#define MASKV_L2 (-1.2984255368e16f)   // -9e15 * log2(e)

typedef __attribute__((ext_vector_type(16))) __bf16 v16bf;
typedef __attribute__((ext_vector_type(8)))  float  v8f;

// ---------------------------------------------------------------------------
// Kernel 1: WhT[n][m] = bf16( (h @ W)[m][n] )   (column-major Wh, bf16)
// One wave per 16-row tile, 4 N-tiles of 16, K=256 in 8 WMMA steps.
// ---------------------------------------------------------------------------
#define WT_RS 264  // padded LDS row stride (bf16 elems), keeps 16B alignment

__global__ __launch_bounds__(256) void gat_wh_kernel(
    const float* __restrict__ h, const float* __restrict__ W,
    __bf16* __restrict__ whT) {
  __shared__ __bf16 WT[64 * WT_RS];
  const int tid = threadIdx.x;
  for (int e = tid; e < FIN * FOUT; e += 256) {
    int k = e >> 6, n = e & 63;          // W row-major [256][64], coalesced read
    WT[n * WT_RS + k] = (__bf16)W[e];
  }
  __syncthreads();

  const int wave = tid >> 5, lane = tid & 31;
  const int m0 = (blockIdx.x * 8 + wave) * 16;
  const int mrow = lane & 15;
  const int hi = lane >> 4;
  const int kbA = hi * 8;     // A-matrix k-group base for this lane half
  const int kbB = hi * 16;    // unused; B loads use kbA, same k order as A
  const float* hrow = h + (size_t)(m0 + mrow) * FIN;

  v8f C[4] = {};
  for (int k0 = 0; k0 < FIN; k0 += 32) {
    float4 a0 = *(const float4*)(hrow + k0 + kbA);
    float4 a1 = *(const float4*)(hrow + k0 + kbA + 4);
    float4 a2 = *(const float4*)(hrow + k0 + kbA + 16);
    float4 a3 = *(const float4*)(hrow + k0 + kbA + 20);
    v16bf A;
    A[0]=(__bf16)a0.x; A[1]=(__bf16)a0.y; A[2]=(__bf16)a0.z; A[3]=(__bf16)a0.w;
    A[4]=(__bf16)a1.x; A[5]=(__bf16)a1.y; A[6]=(__bf16)a1.z; A[7]=(__bf16)a1.w;
    A[8]=(__bf16)a2.x; A[9]=(__bf16)a2.y; A[10]=(__bf16)a2.z; A[11]=(__bf16)a2.w;
    A[12]=(__bf16)a3.x; A[13]=(__bf16)a3.y; A[14]=(__bf16)a3.z; A[15]=(__bf16)a3.w;
#pragma unroll
    for (int t = 0; t < 4; ++t) {
      int n = mrow + 16 * t;
      typedef __attribute__((ext_vector_type(8))) __bf16 v8bf;
      v8bf Blo = *(const v8bf*)&WT[n * WT_RS + k0 + kbA];
      v8bf Bhi = *(const v8bf*)&WT[n * WT_RS + k0 + kbA + 16];
      v16bf B = __builtin_shufflevector(Blo, Bhi, 0,1,2,3,4,5,6,7,8,9,10,11,12,13,14,15);
      C[t] = __builtin_amdgcn_wmma_f32_16x16x32_bf16(
          false, A, false, B, (short)0, C[t], false, false);
    }
  }
#pragma unroll
  for (int t = 0; t < 4; ++t) {
    int n = mrow + 16 * t;
#pragma unroll
    for (int r = 0; r < 8; ++r) {
      int m = m0 + r + 8 * hi;             // C layout: VGPR r -> row r (+8 hi half)
      whT[(size_t)n * NN + m] = (__bf16)C[t][r];
    }
  }
}

// ---------------------------------------------------------------------------
// Kernel 1b: f_src[i] = log2e * Wh[i,:].a[0:64], f_dst[i] = log2e * Wh[i,:].a[64:]
// Pre-scaling by log2(e) moves all softmax exponentials into the exp2 domain
// (leaky_relu is positively homogeneous, so the scaling commutes with it).
// ---------------------------------------------------------------------------
__global__ __launch_bounds__(256) void gat_f_kernel(
    const __bf16* __restrict__ whT, const float* __restrict__ a,
    float* __restrict__ f_src, float* __restrict__ f_dst) {
  int i = blockIdx.x * 256 + threadIdx.x;
  float fs = 0.f, fd = 0.f;
#pragma unroll 8
  for (int n = 0; n < FOUT; ++n) {
    float w = (float)whT[(size_t)n * NN + i];
    fs = fmaf(w, a[n], fs);
    fd = fmaf(w, a[FOUT + n], fd);
  }
  f_src[i] = fs * LOG2E;
  f_dst[i] = fd * LOG2E;
}

__device__ __forceinline__ float gat_score(int av, float fd, float fs) {
  float e = fs + fd;
  e = fmaxf(e, ALPHA * e);       // leaky_relu (alpha < 1), log2 domain
  return av > 0 ? e : MASKV_L2;
}

// ---------------------------------------------------------------------------
// Kernel 2: fused masked-softmax + attention@Wh (flash-style online softmax,
// exp2 domain). One wave owns a 16-row tile and one j-split.
// ---------------------------------------------------------------------------
__global__ __launch_bounds__(128) void gat_attn_kernel(
    const int* __restrict__ adj, const __bf16* __restrict__ whT,
    const float* __restrict__ f_src, const float* __restrict__ f_dst,
    float* __restrict__ partO, float* __restrict__ partM,
    float* __restrict__ partL, int jsplit) {
  const int wave = threadIdx.x >> 5, lane = threadIdx.x & 31;
  const int gw = blockIdx.x * 4 + wave;
  const int tile = gw / jsplit;
  const int split = gw % jsplit;
  const int i0 = tile * 16;
  const int nch = 256 / jsplit;            // 32-col chunks in this split
  const int jstart = split * nch * 32;
  const int jend = jstart + nch * 32;

  const int mrow = lane & 15;
  const int hi = lane >> 4;
  const int kbA = hi * 8;
  const int kbB = hi * 16;
  const int permBase = hi << 3;            // C-layout row base for bpermute

  const int* adjRow = adj + (size_t)(i0 + mrow) * NN;
  const float fs = f_src[i0 + mrow];

  float mrun = -INFINITY;
  float lrun = 0.f;
  v8f O[4] = {};

  // prologue loads (chunk 0)
  int4 A0, A1, A2, A3;
  float4 F0, F1, F2, F3;
  {
    const int jc = jstart + kbA;
    A0 = *(const int4*)(adjRow + jc);      A1 = *(const int4*)(adjRow + jc + 4);
    A2 = *(const int4*)(adjRow + jc + 16); A3 = *(const int4*)(adjRow + jc + 20);
    F0 = *(const float4*)(f_dst + jc);      F1 = *(const float4*)(f_dst + jc + 4);
    F2 = *(const float4*)(f_dst + jc + 16); F3 = *(const float4*)(f_dst + jc + 20);
  }

  int j0 = jstart;
  for (int c = 0; c < nch; ++c) {
    const int jcur = j0;
    int4 cA[4] = {A0, A1, A2, A3};
    float4 cF[4] = {F0, F1, F2, F3};
    // issue next chunk's streaming loads early (register double buffer)
    if (c + 1 < nch) {
      const int jc = j0 + 32 + kbA;
      A0 = *(const int4*)(adjRow + jc);      A1 = *(const int4*)(adjRow + jc + 4);
      A2 = *(const int4*)(adjRow + jc + 16); A3 = *(const int4*)(adjRow + jc + 20);
      F0 = *(const float4*)(f_dst + jc);      F1 = *(const float4*)(f_dst + jc + 4);
      F2 = *(const float4*)(f_dst + jc + 16); F3 = *(const float4*)(f_dst + jc + 20);
    }
    j0 += 32;
    if (jcur + 288 <= jend)                 // deep prefetch of adj stream
      __builtin_prefetch(adjRow + jcur + 256, 0, 3);

    // B fragments: WhT rows contiguous in j (L2-resident, 32B/lane)
    typedef __attribute__((ext_vector_type(8))) __bf16 v8bf;
    const __bf16* bbase = whT + (size_t)mrow * NN + jcur + kbA;
#define LDB(p) __builtin_shufflevector(*(const v8bf*)(p), *(const v8bf*)((p) + 16), 0,1,2,3,4,5,6,7,8,9,10,11,12,13,14,15)
    v16bf B0 = LDB(bbase);
    v16bf B1 = LDB(bbase + (size_t)16 * NN);
    v16bf B2 = LDB(bbase + (size_t)32 * NN);
    v16bf B3 = LDB(bbase + (size_t)48 * NN);
#undef LDB

    // scores (A-fragment element order), log2 domain
    float s[16];
#pragma unroll
    for (int g = 0; g < 4; ++g) {
      s[4*g+0] = gat_score(cA[g].x, cF[g].x, fs);
      s[4*g+1] = gat_score(cA[g].y, cF[g].y, fs);
      s[4*g+2] = gat_score(cA[g].z, cF[g].z, fs);
      s[4*g+3] = gat_score(cA[g].w, cF[g].w, fs);
    }
    // tree max over the lane's 16 scores
    float mx[8];
#pragma unroll
    for (int j = 0; j < 8; ++j) mx[j] = fmaxf(s[j], s[j + 8]);
#pragma unroll
    for (int j = 0; j < 4; ++j) mx[j] = fmaxf(mx[j], mx[j + 4]);
    float cmax = fmaxf(fmaxf(mx[0], mx[1]), fmaxf(mx[2], mx[3]));
    cmax = fmaxf(cmax, __shfl_xor(cmax, 16, 32));   // row split across lane halves
    const float mnew = fmaxf(mrun, cmax);

    // rescale only when some row's max actually advanced (rare once warmed up)
    if (__any(mnew > mrun)) {
      float scl = __builtin_amdgcn_exp2f(mrun - mnew);
      lrun *= scl;
#pragma unroll
      for (int r = 0; r < 8; ++r) {
        float sf = __shfl(scl, permBase + r, 32);
        O[0][r] = O[0][r] * sf;
        O[1][r] = O[1][r] * sf;
        O[2][r] = O[2][r] * sf;
        O[3][r] = O[3][r] * sf;
      }
    }
    mrun = mnew;

    float p[16];
    v16bf P;
#pragma unroll
    for (int j = 0; j < 16; ++j) {
      p[j] = __builtin_amdgcn_exp2f(s[j] - mnew);   // bare v_exp_f32
      P[j] = (__bf16)p[j];
    }
    // tree sum
#pragma unroll
    for (int j = 0; j < 8; ++j) p[j] += p[j + 8];
#pragma unroll
    for (int j = 0; j < 4; ++j) p[j] += p[j + 4];
    float psum = (p[0] + p[1]) + (p[2] + p[3]);
    psum += __shfl_xor(psum, 16, 32);
    lrun += psum;

    O[0] = __builtin_amdgcn_wmma_f32_16x16x32_bf16(false, P, false, B0, (short)0, O[0], false, false);
    O[1] = __builtin_amdgcn_wmma_f32_16x16x32_bf16(false, P, false, B1, (short)0, O[1], false, false);
    O[2] = __builtin_amdgcn_wmma_f32_16x16x32_bf16(false, P, false, B2, (short)0, O[2], false, false);
    O[3] = __builtin_amdgcn_wmma_f32_16x16x32_bf16(false, P, false, B3, (short)0, O[3], false, false);
  }

  // write partials (un-normalized): partO[split][i][n], partM/L[split][i]
#pragma unroll
  for (int r = 0; r < 8; ++r) {
    int i = i0 + r + 8 * hi;
    float* po = partO + ((size_t)split * NN + i) * FOUT;
#pragma unroll
    for (int t = 0; t < 4; ++t) po[mrow + 16 * t] = O[t][r];
  }
  if (lane < 16) {
    partM[(size_t)split * NN + i0 + mrow] = mrun;
    partL[(size_t)split * NN + i0 + mrow] = lrun;
  }
}

// ---------------------------------------------------------------------------
// Kernel 3: merge j-splits (flash merge, exp2 domain), normalize, ELU.
// ---------------------------------------------------------------------------
__global__ __launch_bounds__(256) void gat_merge_kernel(
    const float* __restrict__ partO, const float* __restrict__ partM,
    const float* __restrict__ partL, float* __restrict__ out, int jsplit) {
  int idx = blockIdx.x * 256 + threadIdx.x;   // [0, N*64)
  int i = idx >> 6;
  float gm = -INFINITY;
  for (int s = 0; s < jsplit; ++s)
    gm = fmaxf(gm, partM[(size_t)s * NN + i]);
  float num = 0.f, den = 0.f;
  for (int s = 0; s < jsplit; ++s) {
    float w = __builtin_amdgcn_exp2f(partM[(size_t)s * NN + i] - gm);
    num = fmaf(w, partO[(size_t)s * NN * FOUT + idx], num);
    den = fmaf(w, partL[(size_t)s * NN + i], den);
  }
  float v = num / den;
  out[idx] = v > 0.f ? v : (__expf(v) - 1.f);   // ELU (alpha=1)
}

// ---------------------------------------------------------------------------
extern "C" void kernel_launch(void* const* d_in, const int* in_sizes, int n_in,
                              void* d_out, int out_size, void* d_ws, size_t ws_size,
                              hipStream_t stream) {
  (void)in_sizes; (void)n_in; (void)out_size;
  const float* h   = (const float*)d_in[0];
  const int*   adj = (const int*)d_in[1];
  const float* W   = (const float*)d_in[2];
  const float* a   = (const float*)d_in[3];
  float* out = (float*)d_out;

  char* ws = (char*)d_ws;
  __bf16* whT = (__bf16*)ws;
  size_t off = (size_t)NN * FOUT * 2;            // 1 MB
  float* f_src = (float*)(ws + off); off += (size_t)NN * 4;
  float* f_dst = (float*)(ws + off); off += (size_t)NN * 4;

  const size_t per_split = (size_t)NN * 8 + (size_t)NN * FOUT * 4;
  int S = 1;
  if (ws_size >= off + 8 * per_split)      S = 8;
  else if (ws_size >= off + 4 * per_split) S = 4;
  else if (ws_size >= off + 2 * per_split) S = 2;

  float* partM = (float*)(ws + off); off += (size_t)S * NN * 4;
  float* partL = (float*)(ws + off); off += (size_t)S * NN * 4;
  float* partO = (float*)(ws + off);

  gat_wh_kernel<<<64, 256, 0, stream>>>(h, W, whT);
  gat_f_kernel<<<NN / 256, 256, 0, stream>>>(whT, a, f_src, f_dst);
  gat_attn_kernel<<<128 * S, 128, 0, stream>>>(adj, whT, f_src, f_dst,
                                               partO, partM, partL, S);
  gat_merge_kernel<<<(NN * FOUT) / 256, 256, 0, stream>>>(partO, partM, partL,
                                                          out, S);
}
